// SimplePointTransformerSeg_39857296507166
// MI455X (gfx1250) — matrix operand order repaired and checked by reading, not compiled, not run on hardware
//
#include <hip/hip_runtime.h>
#include <math.h>

// ---------------------------------------------------------------------------
// Point Transformer segmentation forward pass for MI455X (gfx1250, wave32).
// All channel-mixing GEMMs run through v_wmma_f32_16x16x32_f16.
// ---------------------------------------------------------------------------

typedef __attribute__((ext_vector_type(16))) _Float16 v16h;
typedef __attribute__((ext_vector_type(8)))  float    v8f;

#define EPI_RELU_LAST  1   // relu applied after residual add
#define EPI_RELU_FIRST 2   // relu applied before residual add

static constexpr float BN_SC = 0.9999950000374996f;  // 1/sqrt(1+1e-5)
static constexpr int   BATCH = 2;
static constexpr int   NPTS  = 8192;
static constexpr int   CCH   = 64;
static constexpr int   MPTS  = 2048;   // NPTS / STRIDE
static constexpr int   NCLS  = 13;
static constexpr int   KNB   = 16;     // neighbours

// ---------------------------------------------------------------------------
// Fused conv1d/conv2d-as-GEMM:  Y[b,o,n] = epi( scale * sum_c W[o,c]*X[b,c,n] )
// Block = 128 threads (4 waves). Each block computes a 64(o) x 32(col) patch;
// wave w owns o-rows [w*16, w*16+16) and BOTH 16-wide column tiles, so the A
// (weight) fragment is reused across two WMMAs per K-step. KPAD is a template
// parameter so the K loop fully unrolls (2..6 v_wmma per wave, no loop carried
// branches around the matrix ops).
// X tile staged to LDS column-major as f16: sXT[col][K] -> each lane's 16
// B-fragment K-values are two runs of 8 contiguous K (16 bytes each).
// VGPR layouts per CDNA5 ISA 7.12.2:
//   A 16x32 f16 : lane<16 -> row M=lane, K {kc..kc+7} in e0..7, {kc+16..+23} in e8..15
//                 lane>=16 -> same rows, K offset +8 within each half
//   B 32x16 f16 : lane<16 -> col N=lane, K {kc..kc+7} in e0..7, {kc+16..+23} in e8..15 ; lane>=16 -> K +8
//   C/D 16x16 f32: elem r -> row o0 + r + (lane>=16 ? 8:0), col n0 + (lane&15)
// ---------------------------------------------------------------------------
template <int KPAD>
__global__ __launch_bounds__(128)
void conv_wmma_kernel(const float* __restrict__ X, const float* __restrict__ W,
                      const float* __restrict__ bias, const float* __restrict__ resid,
                      float* __restrict__ Y,
                      int Cin, long long Ncols, int Oc, float scale, int flags)
{
    const int b    = blockIdx.z;
    const long long n0 = (long long)blockIdx.x * 32;
    const int wave = threadIdx.x >> 5;
    const int lane = threadIdx.x & 31;
    const int o0   = blockIdx.y * 64 + wave * 16;

    __shared__ _Float16 sXT[32][KPAD + 4];      // column-major activation tile

    const float* Xb = X + (long long)b * Cin * Ncols;
    for (int t = threadIdx.x; t < KPAD * 32; t += 128) {
        const int kk = t >> 5, nn = t & 31;
        const float v = (kk < Cin) ? Xb[(long long)kk * Ncols + n0 + nn] : 0.0f;
        sXT[nn][kk] = (_Float16)v;
    }
    __syncthreads();

    v8f acc0 = {};
    v8f acc1 = {};
    const int  row = lane & 15;
    const bool hi  = lane >= 16;
    const int  o   = o0 + row;

    #pragma unroll
    for (int kc = 0; kc < KPAD; kc += 32) {
        // --- A: weight tile, rows o0..o0+15, K = kc..kc+31 (shared by both cols) ---
        v16h A;
        const int ka = kc + (hi ? 8 : 0);
        #pragma unroll
        for (int j = 0; j < 8; ++j) {
            const int c0 = ka + j;
            const int c1 = ka + 16 + j;
            A[j]     = (o < Oc && c0 < Cin) ? (_Float16)W[(long long)o * Cin + c0] : (_Float16)0.0f;
            A[8 + j] = (o < Oc && c1 < Cin) ? (_Float16)W[(long long)o * Cin + c1] : (_Float16)0.0f;
        }
        // --- B fragments for the two 16-wide column tiles ---
        const int kb = kc + (hi ? 8 : 0);
        v16h B0, B1;
        #pragma unroll
        for (int e = 0; e < 16; ++e) {
            B0[e] = sXT[row][kb + e + (e >= 8 ? 8 : 0)];
            B1[e] = sXT[16 + row][kb + e + (e >= 8 ? 8 : 0)];
        }
        acc0 = __builtin_amdgcn_wmma_f32_16x16x32_f16(false, A, false, B0, (short)0, acc0, false, false);
        acc1 = __builtin_amdgcn_wmma_f32_16x16x32_f16(false, A, false, B1, (short)0, acc1, false, false);
    }

    // --- epilogue + store (two tiles) ---
    #pragma unroll
    for (int t = 0; t < 2; ++t) {
        const v8f& acc = t ? acc1 : acc0;
        const long long n = n0 + t * 16 + row;
        #pragma unroll
        for (int r = 0; r < 8; ++r) {
            const int oo = o0 + r + (hi ? 8 : 0);
            if (oo < Oc) {
                float v = scale * acc[r];
                if (bias) v += bias[oo];
                if (flags & EPI_RELU_FIRST) v = fmaxf(v, 0.0f);
                if (resid) v += resid[((long long)b * Oc + oo) * Ncols + n];
                if (flags & EPI_RELU_LAST)  v = fmaxf(v, 0.0f);
                Y[((long long)b * Oc + oo) * Ncols + n] = v;
            }
        }
    }
}

// ---------------------------------------------------------------------------
// kNN: one thread per query keeps a sorted top-K (ascending distance, ties ->
// earlier ref index, matching jax.lax.top_k on -d).
// ---------------------------------------------------------------------------
template <int KN>
__global__ void knn_kernel(const float* __restrict__ qpts, const float* __restrict__ rpts,
                           int* __restrict__ oidx, float* __restrict__ odist,
                           int Nq, int Nr)
{
    const int b = blockIdx.y;
    const int i = blockIdx.x * blockDim.x + threadIdx.x;
    if (i >= Nq) return;
    const float* Q = qpts + ((long long)b * Nq + i) * 3;
    const float* R = rpts + (long long)b * Nr * 3;
    const float qx = Q[0], qy = Q[1], qz = Q[2];
    const float qq = qx * qx + qy * qy + qz * qz;

    float bd[KN]; int bi[KN];
    #pragma unroll
    for (int k = 0; k < KN; ++k) { bd[k] = 3.4e38f; bi[k] = 0; }

    for (int j = 0; j < Nr; ++j) {
        if ((j & 31) == 0 && j + 64 < Nr)
            __builtin_prefetch(R + 3 * (j + 64), 0, 0);   // global_prefetch_b8
        const float rx = R[3 * j], ry = R[3 * j + 1], rz = R[3 * j + 2];
        const float d = qq + rx * rx + ry * ry + rz * rz
                        - 2.0f * (qx * rx + qy * ry + qz * rz);
        if (d < bd[KN - 1]) {
            int p = KN - 1;
            while (p > 0 && d < bd[p - 1]) { bd[p] = bd[p - 1]; bi[p] = bi[p - 1]; --p; }
            bd[p] = d; bi[p] = j;
        }
    }
    #pragma unroll
    for (int k = 0; k < KN; ++k) {
        oidx [((long long)b * Nq + i) * KN + k] = bi[k];
        odist[((long long)b * Nq + i) * KN + k] = fmaxf(bd[k], 0.0f);
    }
}

// ---------------------------------------------------------------------------
// Farthest point sampling: one 1024-thread workgroup per batch, distances in
// registers (8192/1024 = 8 per thread), LDS tree argmax (tie -> lowest index).
// ---------------------------------------------------------------------------
__global__ __launch_bounds__(1024)
void fps_kernel(const float* __restrict__ xyz, int* __restrict__ sel,
                float* __restrict__ pout)
{
    const int b = blockIdx.x, tid = threadIdx.x;
    const float* P = xyz + (long long)b * NPTS * 3;
    __shared__ float sv[1024];
    __shared__ int   si[1024];
    __shared__ float spt[3];

    const float x0 = P[0], y0 = P[1], z0 = P[2];
    float d[8];
    #pragma unroll
    for (int e = 0; e < 8; ++e) {
        const int i = tid + e * 1024;
        const float dx = P[3 * i] - x0, dy = P[3 * i + 1] - y0, dz = P[3 * i + 2] - z0;
        d[e] = dx * dx + dy * dy + dz * dz;
    }
    if (tid == 0) {
        sel[(long long)b * MPTS] = 0;
        pout[(long long)b * MPTS * 3 + 0] = x0;
        pout[(long long)b * MPTS * 3 + 1] = y0;
        pout[(long long)b * MPTS * 3 + 2] = z0;
    }

    for (int it = 1; it < MPTS; ++it) {
        float bv = -1.0f; int bi = 0;
        #pragma unroll
        for (int e = 0; e < 8; ++e) {
            const int i = tid + e * 1024;
            if (d[e] > bv) { bv = d[e]; bi = i; }   // strict > keeps earliest i
        }
        sv[tid] = bv; si[tid] = bi;
        __syncthreads();
        for (int s = 512; s > 0; s >>= 1) {
            if (tid < s) {
                const float v2 = sv[tid + s]; const int j2 = si[tid + s];
                if (v2 > sv[tid] || (v2 == sv[tid] && j2 < si[tid])) { sv[tid] = v2; si[tid] = j2; }
            }
            __syncthreads();
        }
        const int nxt = si[0];
        if (tid == 0) {
            sel[(long long)b * MPTS + it] = nxt;
            spt[0] = P[3 * nxt]; spt[1] = P[3 * nxt + 1]; spt[2] = P[3 * nxt + 2];
            pout[((long long)b * MPTS + it) * 3 + 0] = spt[0];
            pout[((long long)b * MPTS + it) * 3 + 1] = spt[1];
            pout[((long long)b * MPTS + it) * 3 + 2] = spt[2];
        }
        __syncthreads();
        const float nx = spt[0], ny = spt[1], nz = spt[2];
        #pragma unroll
        for (int e = 0; e < 8; ++e) {
            const int i = tid + e * 1024;
            const float dx = P[3 * i] - nx, dy = P[3 * i + 1] - ny, dz = P[3 * i + 2] - nz;
            d[e] = fminf(d[e], dx * dx + dy * dy + dz * dz);
        }
        __syncthreads();
    }
}

// ---------------------------------------------------------------------------
// z1[c, n*16+k] = relu(BN * (q[c,n] - ke[c,idx] + r[c,n,k]))
// where r = P2 @ relu(BN * (P1 @ rel)) + bP2.  One 64-thread block per (b,n).
// ---------------------------------------------------------------------------
__global__ __launch_bounds__(64)
void build_z1_kernel(const float* __restrict__ xyz, const int* __restrict__ idx,
                     const float* __restrict__ q, const float* __restrict__ ke,
                     const float* __restrict__ P1, const float* __restrict__ P2,
                     const float* __restrict__ bP2,
                     float* __restrict__ z1, int Np)
{
    const int b = blockIdx.y, n = blockIdx.x, c = threadIdx.x;
    __shared__ float t[16][3];
    __shared__ int   sI[16];
    const float* Pb = xyz + (long long)b * Np * 3;

    if (c < 16) {
        const int iv = idx[((long long)b * Np + n) * KNB + c];
        sI[c] = iv;
        const float r0 = Pb[3 * iv]     - Pb[3 * n];
        const float r1 = Pb[3 * iv + 1] - Pb[3 * n + 1];
        const float r2 = Pb[3 * iv + 2] - Pb[3 * n + 2];
        #pragma unroll
        for (int j = 0; j < 3; ++j) {
            const float s = P1[j * 3] * r0 + P1[j * 3 + 1] * r1 + P1[j * 3 + 2] * r2;
            t[c][j] = fmaxf(BN_SC * s, 0.0f);
        }
    }
    __syncthreads();

    const long long Ncols = (long long)Np * KNB;
    const float qc = q[((long long)b * CCH + c) * Np + n];
    const float p20 = P2[c * 3], p21 = P2[c * 3 + 1], p22 = P2[c * 3 + 2], bp = bP2[c];
    #pragma unroll
    for (int k = 0; k < KNB; ++k) {
        const float r  = bp + p20 * t[k][0] + p21 * t[k][1] + p22 * t[k][2];
        const float kv = ke[((long long)b * CCH + c) * Np + sI[k]];
        const float a  = qc - kv + r;
        z1[((long long)b * CCH + c) * Ncols + (long long)n * KNB + k] = fmaxf(BN_SC * a, 0.0f);
    }
}

// ---------------------------------------------------------------------------
// out[c,n] = epi( sum_k softmax_k(a[c,n,:]) * (v[c,idx[n,k]] + r[c,n,k]) )
// ---------------------------------------------------------------------------
__global__ __launch_bounds__(64)
void attn_final_kernel(const float* __restrict__ xyz, const int* __restrict__ idx,
                       const float* __restrict__ a, const float* __restrict__ v,
                       const float* __restrict__ P1, const float* __restrict__ P2,
                       const float* __restrict__ bP2,
                       float* __restrict__ out, int Np, float outscale, int dorelu)
{
    const int b = blockIdx.y, n = blockIdx.x, c = threadIdx.x;
    __shared__ float t[16][3];
    __shared__ int   sI[16];
    const float* Pb = xyz + (long long)b * Np * 3;

    if (c < 16) {
        const int iv = idx[((long long)b * Np + n) * KNB + c];
        sI[c] = iv;
        const float r0 = Pb[3 * iv]     - Pb[3 * n];
        const float r1 = Pb[3 * iv + 1] - Pb[3 * n + 1];
        const float r2 = Pb[3 * iv + 2] - Pb[3 * n + 2];
        #pragma unroll
        for (int j = 0; j < 3; ++j) {
            const float s = P1[j * 3] * r0 + P1[j * 3 + 1] * r1 + P1[j * 3 + 2] * r2;
            t[c][j] = fmaxf(BN_SC * s, 0.0f);
        }
    }
    __syncthreads();

    const long long Ncols = (long long)Np * KNB;
    float av[KNB];
    #pragma unroll
    for (int k = 0; k < KNB; ++k)
        av[k] = a[((long long)b * CCH + c) * Ncols + (long long)n * KNB + k];

    float m = av[0];
    #pragma unroll
    for (int k = 1; k < KNB; ++k) m = fmaxf(m, av[k]);
    float s = 0.0f;
    #pragma unroll
    for (int k = 0; k < KNB; ++k) { av[k] = expf(av[k] - m); s += av[k]; }
    const float inv = 1.0f / s;

    const float p20 = P2[c * 3], p21 = P2[c * 3 + 1], p22 = P2[c * 3 + 2], bp = bP2[c];
    float acc = 0.0f;
    #pragma unroll
    for (int k = 0; k < KNB; ++k) {
        const float r  = bp + p20 * t[k][0] + p21 * t[k][1] + p22 * t[k][2];
        const float vv = v[((long long)b * CCH + c) * Np + sI[k]];
        acc += (vv + r) * av[k] * inv;
    }
    float o = outscale * acc;
    if (dorelu) o = fmaxf(o, 0.0f);
    out[((long long)b * CCH + c) * Np + n] = o;
}

// ---------------------------------------------------------------------------
// Misc elementwise kernels
// ---------------------------------------------------------------------------
__global__ void transpose_xyz_kernel(const float* __restrict__ xyz, float* __restrict__ X0)
{
    const long long i = (long long)blockIdx.x * 256 + threadIdx.x;
    const long long total = (long long)BATCH * 3 * NPTS;
    if (i >= total) return;
    const int n = (int)(i % NPTS);
    long long r = i / NPTS;
    const int c = (int)(r % 3);
    const int b = (int)(r / 3);
    X0[i] = xyz[((long long)b * NPTS + n) * 3 + c];
}

// nx[b, c, m*16+k] : c<3 -> rel, else gathered feat
__global__ void build_nx_kernel(const float* __restrict__ xyz, const float* __restrict__ pout,
                                const int* __restrict__ idx, const float* __restrict__ feat,
                                float* __restrict__ nx)
{
    const long long i = (long long)blockIdx.x * 256 + threadIdx.x;
    const long long total = (long long)BATCH * 67 * MPTS * KNB;
    if (i >= total) return;
    const int k = (int)(i % KNB);
    long long r = i / KNB;
    const int m = (int)(r % MPTS); r /= MPTS;
    const int c = (int)(r % 67);
    const int b = (int)(r / 67);
    const int iv = idx[((long long)b * MPTS + m) * KNB + k];
    float val;
    if (c < 3)
        val = xyz[((long long)b * NPTS + iv) * 3 + c] - pout[((long long)b * MPTS + m) * 3 + c];
    else
        val = feat[((long long)b * CCH + (c - 3)) * NPTS + iv];
    nx[((long long)b * 67 + c) * (MPTS * KNB) + (long long)m * KNB + k] = val;
}

__global__ void maxk_kernel(const float* __restrict__ z, float* __restrict__ out)
{
    const long long i = (long long)blockIdx.x * 256 + threadIdx.x;
    const long long total = (long long)BATCH * CCH * MPTS;
    if (i >= total) return;
    const int m = (int)(i % MPTS);
    long long r = i / MPTS;
    const int c = (int)(r % CCH);
    const int b = (int)(r / CCH);
    const float* p = z + ((long long)b * CCH + c) * (MPTS * KNB) + (long long)m * KNB;
    float v = p[0];
    #pragma unroll
    for (int k = 1; k < KNB; ++k) v = fmaxf(v, p[k]);
    out[i] = v;
}

__global__ void interp_up_kernel(const float* __restrict__ f1, const int* __restrict__ idx3,
                                 const float* __restrict__ d3, float* __restrict__ up)
{
    const long long i = (long long)blockIdx.x * 256 + threadIdx.x;
    const long long total = (long long)BATCH * CCH * NPTS;
    if (i >= total) return;
    const int n = (int)(i % NPTS);
    long long r = i / NPTS;
    const int c = (int)(r % CCH);
    const int b = (int)(r / CCH);
    const long long base = ((long long)b * NPTS + n) * 3;
    const float w0 = 1.0f / (d3[base]     + 1e-8f);
    const float w1 = 1.0f / (d3[base + 1] + 1e-8f);
    const float w2 = 1.0f / (d3[base + 2] + 1e-8f);
    const float ws = w0 + w1 + w2;
    const int i0 = idx3[base], i1 = idx3[base + 1], i2 = idx3[base + 2];
    const float* f = f1 + ((long long)b * CCH + c) * MPTS;
    up[i] = (f[i0] * w0 + f[i1] * w1 + f[i2] * w2) / ws;
}

// ---------------------------------------------------------------------------
// Host-side orchestration
// ---------------------------------------------------------------------------
struct BlkP {
    const float *A1, *A2, *P1, *P2, *W1, *W2, *Wk, *Wq, *Wv, *bA2, *bP2, *bk, *bq, *bv;
};

extern "C" void kernel_launch(void* const* d_in, const int* in_sizes, int n_in,
                              void* d_out, int out_size, void* d_ws, size_t ws_size,
                              hipStream_t stream)
{
    (void)in_sizes; (void)n_in; (void)out_size; (void)ws_size;

    // JAX pytree leaf order: sorted dict keys (ASCII: uppercase < lowercase).
    // top: inputs, then params: O1 O2 bO2 block1 block2 block3 down in1 in2 up
    // block: A1 A2 P1 P2 W1 W2 Wk Wq Wv bA2 bP2 bk bq bv ; down: D1 D2 ; up: U1 U2
    int p = 0;
    auto F = [&](void) -> const float* { return (const float*)d_in[p++]; };
    const float* xyz = F();
    const float* O1 = F(); const float* O2 = F(); const float* bO2 = F();
    auto RB = [&](void) -> BlkP {
        BlkP r;
        r.A1 = F(); r.A2 = F(); r.P1 = F(); r.P2 = F(); r.W1 = F(); r.W2 = F();
        r.Wk = F(); r.Wq = F(); r.Wv = F(); r.bA2 = F(); r.bP2 = F(); r.bk = F();
        r.bq = F(); r.bv = F();
        return r;
    };
    const BlkP B1 = RB(), B2 = RB(), B3 = RB();
    const float* D1w = F(); const float* D2w = F();
    const float* in1 = F(); const float* in2 = F();
    const float* U1w = F(); const float* U2w = F();

    // ---- workspace bump allocator ----
    char* wp = (char*)d_ws;
    auto allocF = [&](long long elems) -> float* {
        float* r = (float*)wp; wp += (size_t)elems * sizeof(float); return r;
    };
    auto allocI = [&](long long elems) -> int* {
        int* r = (int*)wp; wp += (size_t)elems * sizeof(int); return r;
    };
    const long long BCN = (long long)BATCH * CCH * NPTS;       // 1,048,576
    const long long BCM = (long long)BATCH * CCH * MPTS;       // 262,144
    const long long BGN = (long long)BATCH * CCH * NPTS * KNB; // 16,777,216

    float* X0  = allocF((long long)BATCH * 3 * NPTS);
    float* F0  = allocF(BCN);  // stem out / block3 out
    float* T0  = allocF(BCN);  // stem tmp / transformer out
    float* Y0  = allocF(BCN);  // block temp y1 / head temp
    float* Qb  = allocF(BCN);
    float* Kb  = allocF(BCN);
    float* Vb  = allocF(BCN);
    float* X1  = allocF(BCN);  // block1 out
    float* UP  = allocF(BCN);
    float* XU  = allocF(BCN);  // transition_up out
    float* P4  = allocF((long long)BATCH * MPTS * 3);
    float* X4A = allocF(BCM);
    float* X4B = allocF(BCM);
    float* F1U = allocF(BCM);
    float* NX  = allocF((long long)BATCH * 67 * MPTS * KNB);
    float* G1  = allocF(BGN);  // big column tensor (ping)
    float* G2  = allocF(BGN);  // big column tensor (pong)
    int*   IdxA = allocI((long long)BATCH * NPTS * KNB);  float* DstA = allocF((long long)BATCH * NPTS * KNB);
    int*   IdxD = allocI((long long)BATCH * MPTS * KNB);  float* DstD = allocF((long long)BATCH * MPTS * KNB);
    int*   IdxB = allocI((long long)BATCH * MPTS * KNB);  float* DstB = allocF((long long)BATCH * MPTS * KNB);
    int*   Idx3 = allocI((long long)BATCH * NPTS * 3);    float* Dst3 = allocF((long long)BATCH * NPTS * 3);
    int*   Sel  = allocI((long long)BATCH * MPTS);

    auto conv = [&](const float* Xp, const float* Wp, const float* bp, const float* rp,
                    float* Yp, int Cin, long long Ncols, int Oc, float scale, int flags) {
        dim3 g((unsigned)(Ncols / 32), (unsigned)((Oc + 63) / 64), BATCH);
        const int kpad = (Cin + 31) & ~31;
        if (kpad == 32)
            conv_wmma_kernel<32><<<g, 128, 0, stream>>>(Xp, Wp, bp, rp, Yp, Cin, Ncols, Oc, scale, flags);
        else if (kpad == 64)
            conv_wmma_kernel<64><<<g, 128, 0, stream>>>(Xp, Wp, bp, rp, Yp, Cin, Ncols, Oc, scale, flags);
        else
            conv_wmma_kernel<96><<<g, 128, 0, stream>>>(Xp, Wp, bp, rp, Yp, Cin, Ncols, Oc, scale, flags);
    };
    auto knn16 = [&](const float* qp, int Nq, const float* rp, int Nr, int* oi, float* od) {
        dim3 g((unsigned)((Nq + 127) / 128), BATCH);
        knn_kernel<16><<<g, 128, 0, stream>>>(qp, rp, oi, od, Nq, Nr);
    };
    auto ew = [&](long long total) -> dim3 { return dim3((unsigned)((total + 255) / 256)); };

    auto transformer = [&](const BlkP& P, const float* xyzp, int Np, const int* idxp,
                           const float* feat, float* out, float outscale, int dorelu) {
        conv(feat, P.Wq, P.bq, nullptr, Qb, CCH, Np, CCH, 1.0f, 0);
        conv(feat, P.Wk, P.bk, nullptr, Kb, CCH, Np, CCH, 1.0f, 0);
        conv(feat, P.Wv, P.bv, nullptr, Vb, CCH, Np, CCH, 1.0f, 0);
        dim3 gz((unsigned)Np, BATCH);
        build_z1_kernel<<<gz, 64, 0, stream>>>(xyzp, idxp, Qb, Kb, P.P1, P.P2, P.bP2, G1, Np);
        conv(G1, P.A1, nullptr, nullptr, G2, CCH, (long long)Np * KNB, CCH, BN_SC, EPI_RELU_LAST);
        conv(G2, P.A2, P.bA2, nullptr, G1, CCH, (long long)Np * KNB, CCH, 1.0f, 0);
        attn_final_kernel<<<gz, 64, 0, stream>>>(xyzp, idxp, G1, Vb, P.P1, P.P2, P.bP2,
                                                 out, Np, outscale, dorelu);
    };
    auto blockf = [&](const BlkP& P, const float* xyzp, int Np, const int* idxp,
                      const float* feat, float* out) {
        conv(feat, P.W1, nullptr, nullptr, Y0, CCH, Np, CCH, BN_SC, EPI_RELU_LAST); // y1
        transformer(P, xyzp, Np, idxp, Y0, T0, BN_SC, 1);                           // relu(bn(tl))
        conv(T0, P.W2, nullptr, feat, out, CCH, Np, CCH, BN_SC, EPI_RELU_LAST);     // relu(bn(.)+feat)
    };

    // ---- stem ----
    transpose_xyz_kernel<<<ew((long long)BATCH * 3 * NPTS), 256, 0, stream>>>(xyz, X0);
    conv(X0, in1, nullptr, nullptr, T0, 3, NPTS, CCH, BN_SC, EPI_RELU_LAST);
    conv(T0, in2, nullptr, nullptr, F0, CCH, NPTS, CCH, BN_SC, EPI_RELU_LAST);

    // ---- block1 ----
    knn16(xyz, NPTS, xyz, NPTS, IdxA, DstA);
    blockf(B1, xyz, NPTS, IdxA, F0, X1);

    // ---- transition down ----
    fps_kernel<<<BATCH, 1024, 0, stream>>>(xyz, Sel, P4);
    knn16(P4, MPTS, xyz, NPTS, IdxD, DstD);
    build_nx_kernel<<<ew((long long)BATCH * 67 * MPTS * KNB), 256, 0, stream>>>(xyz, P4, IdxD, X1, NX);
    conv(NX, D1w, nullptr, nullptr, G1, 67, (long long)MPTS * KNB, CCH, BN_SC, EPI_RELU_LAST);
    conv(G1, D2w, nullptr, nullptr, G2, CCH, (long long)MPTS * KNB, CCH, BN_SC, EPI_RELU_LAST);
    maxk_kernel<<<ew(BCM), 256, 0, stream>>>(G2, X4A);

    // ---- block2 ----
    knn16(P4, MPTS, P4, MPTS, IdxB, DstB);
    blockf(B2, P4, MPTS, IdxB, X4A, X4B);

    // ---- transition up ----
    {
        dim3 g((unsigned)((NPTS + 127) / 128), BATCH);
        knn_kernel<3><<<g, 128, 0, stream>>>(xyz, P4, Idx3, Dst3, NPTS, MPTS);
    }
    conv(X4B, U1w, nullptr, nullptr, F1U, CCH, MPTS, CCH, BN_SC, EPI_RELU_LAST);   // f1
    interp_up_kernel<<<ew(BCN), 256, 0, stream>>>(F1U, Idx3, Dst3, UP);
    conv(X1, U2w, nullptr, UP, XU, CCH, NPTS, CCH, BN_SC, EPI_RELU_FIRST);         // relu(bn(.)) + up

    // ---- block3 (p == xyz, so kNN identical to block1's) ----
    blockf(B3, xyz, NPTS, IdxA, XU, F0);

    // ---- head ----
    conv(F0, O1, nullptr, nullptr, Y0, CCH, NPTS, CCH, BN_SC, EPI_RELU_LAST);
    conv(Y0, O2, bO2, nullptr, (float*)d_out, CCH, NPTS, NCLS, 1.0f, 0);
}
